// WindowAttention_63161789055021
// MI455X (gfx1250) — matrix operand order repaired and checked by reading, not compiled, not run on hardware
//
#include <hip/hip_runtime.h>
#include <hip/hip_bf16.h>

typedef _Float16 half4_t  __attribute__((ext_vector_type(4)));
typedef _Float16 half8_t  __attribute__((ext_vector_type(8)));
typedef _Float16 half16_t __attribute__((ext_vector_type(16)));
typedef float    f32x8    __attribute__((ext_vector_type(8)));

static __device__ inline half16_t cat8(half8_t lo, half8_t hi) {
  return __builtin_shufflevector(lo, hi, 0,1,2,3,4,5,6,7,8,9,10,11,12,13,14,15);
}
static __device__ inline f32x8 wmma_f16(half16_t a, half16_t b, f32x8 c) {
  // D = A*B + C ; f16 inputs, f32 accumulate -> v_wmma_f32_16x16x32_f16
  return __builtin_amdgcn_wmma_f32_16x16x32_f16(false, a, false, b, (short)0, c, false, false);
}

// ---------------------------------------------------------------------------
// Kernel 1: LayerNorm + window partition, fp32 -> f16 (window-major tokens)
// x: (32, 3136, 192) fp32 ; xw: (100352, 192) f16, row t = w*49+p
// ---------------------------------------------------------------------------
__global__ __launch_bounds__(256) void ln_partition(const float* __restrict__ x,
                                                    const float* __restrict__ gamma,
                                                    _Float16* __restrict__ xw) {
  const int tid  = threadIdx.x;
  const int wv   = tid >> 5;
  const int lane = tid & 31;
  const int t = blockIdx.x * 8 + wv;           // token in window order, < 100352

  const int w   = t / 49;
  const int p   = t - w * 49;
  const int b   = w >> 6;
  const int win = w & 63;
  const int wh  = win >> 3, ww = win & 7;
  const int pi  = p / 7,    pj = p - pi * 7;
  const int norg = (wh * 7 + pi) * 56 + (ww * 7 + pj);
  const float* src = x + ((size_t)b * 3136 + norg) * 192;

  float vals[6];
  float s = 0.f, s2 = 0.f;
#pragma unroll
  for (int i = 0; i < 6; ++i) {
    float v = src[lane + 32 * i];
    vals[i] = v; s += v; s2 += v * v;
  }
#pragma unroll
  for (int msk = 16; msk >= 1; msk >>= 1) {
    s  += __shfl_xor(s,  msk, 32);
    s2 += __shfl_xor(s2, msk, 32);
  }
  const float mu  = s * (1.0f / 192.0f);
  const float var = s2 * (1.0f / 192.0f) - mu * mu;
  const float rs  = rsqrtf(var + 1e-5f);
#pragma unroll
  for (int i = 0; i < 6; ++i) {
    int c = lane + 32 * i;
    xw[(size_t)t * 192 + c] = (_Float16)((vals[i] - mu) * rs * gamma[c]);
  }
}

// ---------------------------------------------------------------------------
// Kernel 2: QKV GEMM  (100352 x 576) = xw (100352x192) * qkv_w^T (192x576) + b
// Scatters into q/k/v: (window, head, token, 32) f16 ; q pre-scaled by d^-0.5
// ---------------------------------------------------------------------------
__global__ __launch_bounds__(256) void qkv_gemm(const _Float16* __restrict__ xw,
                                                const float* __restrict__ Wg,
                                                const float* __restrict__ bg,
                                                _Float16* __restrict__ q,
                                                _Float16* __restrict__ k,
                                                _Float16* __restrict__ v) {
  __shared__ __align__(16) _Float16 As[128 * 200];   // A tile, padded stride
  __shared__ __align__(16) _Float16 Bt[64 * 200];    // B^T tile (rows = N cols)
  const int tid = threadIdx.x;
  const int m0 = blockIdx.x * 128;
  const int n0 = blockIdx.y * 64;

  for (int i = tid; i < 128 * 24; i += 256) {        // A: 128 rows x 192 f16
    int m = i / 24, kc = i % 24;
    *(uint4*)(&As[m * 200 + kc * 8]) = *(const uint4*)(xw + (size_t)(m0 + m) * 192 + kc * 8);
  }
  for (int i = tid; i < 64 * 48; i += 256) {         // Bt row n = W row (n0+n)
    int n = i / 48, kc = i % 48;
    const float4 w4 = *(const float4*)(Wg + (size_t)(n0 + n) * 192 + kc * 4);
    half4_t h4 = { (_Float16)w4.x, (_Float16)w4.y, (_Float16)w4.z, (_Float16)w4.w };
    *(half4_t*)(&Bt[n * 200 + kc * 4]) = h4;
  }
  __syncthreads();

  const int wv = tid >> 5;
  const int lane = tid & 31;
  const int lh = lane & 15;
  const bool hiH = lane >= 16;
  const int mb = wv * 16;

  f32x8 acc[4] = {};
#pragma unroll
  for (int ks = 0; ks < 6; ++ks) {                   // K = 6 * 32
    const _Float16* ap = &As[(mb + lh) * 200 + ks * 32 + (hiH ? 16 : 0)];
    half16_t aop = cat8(*(const half8_t*)ap, *(const half8_t*)(ap + 8));
#pragma unroll
    for (int ni = 0; ni < 4; ++ni) {
      const _Float16* bp = &Bt[(ni * 16 + lh) * 200 + ks * 32 + (hiH ? 16 : 0)];
      half16_t bop = cat8(*(const half8_t*)bp, *(const half8_t*)(bp + 8));
      acc[ni] = wmma_f16(aop, bop, acc[ni]);
    }
  }

  const float qscale = 0.17677669529663687f;         // 32^-0.5
#pragma unroll
  for (int ni = 0; ni < 4; ++ni) {
    int j = n0 + ni * 16 + lh;
    int which = j / 192;
    int jh = j - which * 192;
    int hh = jh >> 5, f = jh & 31;
    float bj = bg[j];
    _Float16* dst = (which == 0) ? q : ((which == 1) ? k : v);
    float sc = (which == 0) ? qscale : 1.0f;
#pragma unroll
    for (int r = 0; r < 8; ++r) {
      int m = m0 + mb + (hiH ? r + 8 : r);
      int wdx = m / 49;
      int p = m - wdx * 49;
      dst[((size_t)(wdx * 6 + hh) * 49 + p) * 32 + f] = (_Float16)((acc[ni][r] + bj) * sc);
    }
  }
}

// ---------------------------------------------------------------------------
// Kernel 3: window attention, one wave per (window, head).
// S = q k^T (+rel-pos bias), softmax, O = P v ; 49 tokens padded to 64.
// Writes attn-out (100352 x 192) f16 with channel = h*32+d.
// ---------------------------------------------------------------------------
__global__ __launch_bounds__(32) void attn_win(const _Float16* __restrict__ qg,
                                               const _Float16* __restrict__ kg,
                                               const _Float16* __restrict__ vg,
                                               const float* __restrict__ rpb,
                                               _Float16* __restrict__ og) {
  const int w = blockIdx.x;
  const int h = blockIdx.y;
  const int lane = threadIdx.x;
  const int lh = lane & 15;
  const bool hiH = lane >= 16;

  __shared__ __align__(16) float     bsh[64 * 64];   // bias, -1e9 in padding
  __shared__ __align__(16) _Float16  vtsh[32 * 64];  // v transposed, token-padded
  __shared__ __align__(16) _Float16  pbuf[16 * 64];  // probability row-block

  const size_t base = (size_t)(w * 6 + h) * (49 * 32);
  const _Float16* qb = qg + base;
  const _Float16* kb = kg + base;
  const _Float16* vb = vg + base;

  for (int i = lane; i < 64 * 64; i += 32) {
    int r = i >> 6, c = i & 63;
    float bv = -1e9f;
    if (r < 49 && c < 49) {
      int ri = r / 7, rj = r - ri * 7;
      int ci = c / 7, cj = c - ci * 7;
      int idx = (ri - ci + 6) * 13 + (rj - cj + 6);
      bv = rpb[idx * 6 + h];
    }
    bsh[i] = bv;
  }
  for (int i = lane; i < 32 * 64; i += 32) {         // vtsh[f][p] = v[p][f]
    int f = i >> 6, p = i & 63;
    vtsh[i] = (p < 49) ? vb[p * 32 + f] : (_Float16)0.f;
  }
  __syncthreads();

  // K as WMMA B operand (lane = key token within n-tile)
  half16_t kop[4];
#pragma unroll
  for (int ni = 0; ni < 4; ++ni) {
    int tok = ni * 16 + lh;
    half8_t z = {}, b0 = z, b1 = z;
    if (tok < 49) {
      const _Float16* kp = kb + tok * 32 + (hiH ? 16 : 0);
      b0 = *(const half8_t*)kp;
      b1 = *(const half8_t*)(kp + 8);
    }
    kop[ni] = cat8(b0, b1);
  }

  for (int mi = 0; mi < 4; ++mi) {
    // Q as WMMA A operand (lane = query token within m-tile)
    half16_t qop;
    {
      int tok = mi * 16 + lh;
      half8_t z = {}, a0 = z, a1 = z;
      if (tok < 49) {
        const _Float16* qp = qb + tok * 32 + (hiH ? 16 : 0);
        a0 = *(const half8_t*)qp;
        a1 = *(const half8_t*)(qp + 8);
      }
      qop = cat8(a0, a1);
    }
    f32x8 s[4];
#pragma unroll
    for (int ni = 0; ni < 4; ++ni) {
      f32x8 cz = {};
      s[ni] = wmma_f16(qop, kop[ni], cz);
    }
    // bias + exact softmax; each half-wave owns 8 rows, 64 cols across lanes
#pragma unroll
    for (int r = 0; r < 8; ++r) {
      int lrow = hiH ? (r + 8) : r;
      int grow = mi * 16 + lrow;
      float v0 = s[0][r] + bsh[grow * 64 +  0 + lh];
      float v1 = s[1][r] + bsh[grow * 64 + 16 + lh];
      float v2 = s[2][r] + bsh[grow * 64 + 32 + lh];
      float v3 = s[3][r] + bsh[grow * 64 + 48 + lh];
      float mx = fmaxf(fmaxf(v0, v1), fmaxf(v2, v3));
#pragma unroll
      for (int msk = 8; msk >= 1; msk >>= 1) mx = fmaxf(mx, __shfl_xor(mx, msk, 32));
      float e0 = __expf(v0 - mx), e1 = __expf(v1 - mx);
      float e2 = __expf(v2 - mx), e3 = __expf(v3 - mx);
      float sm = e0 + e1 + e2 + e3;
#pragma unroll
      for (int msk = 8; msk >= 1; msk >>= 1) sm += __shfl_xor(sm, msk, 32);
      float inv = __builtin_amdgcn_rcpf(sm);
      pbuf[lrow * 64 +  0 + lh] = (_Float16)(e0 * inv);
      pbuf[lrow * 64 + 16 + lh] = (_Float16)(e1 * inv);
      pbuf[lrow * 64 + 32 + lh] = (_Float16)(e2 * inv);
      pbuf[lrow * 64 + 48 + lh] = (_Float16)(e3 * inv);
    }
    __syncthreads();

    f32x8 o[2] = {};
#pragma unroll
    for (int kc = 0; kc < 2; ++kc) {
      const _Float16* pp = &pbuf[lh * 64 + kc * 32 + (hiH ? 16 : 0)];
      half16_t pop = cat8(*(const half8_t*)pp, *(const half8_t*)(pp + 8));
#pragma unroll
      for (int nj = 0; nj < 2; ++nj) {
        const _Float16* vp = &vtsh[(nj * 16 + lh) * 64 + kc * 32 + (hiH ? 16 : 0)];
        half16_t vop = cat8(*(const half8_t*)vp, *(const half8_t*)(vp + 8));
        o[nj] = wmma_f16(pop, vop, o[nj]);
      }
    }
#pragma unroll
    for (int r = 0; r < 8; ++r) {
      int lrow = hiH ? (r + 8) : r;
      int tok = mi * 16 + lrow;
      if (tok < 49) {
        size_t ob = ((size_t)w * 49 + tok) * 192 + h * 32;
        og[ob + lh]      = (_Float16)o[0][r];
        og[ob + 16 + lh] = (_Float16)o[1][r];
      }
    }
    __syncthreads();
  }
}

// ---------------------------------------------------------------------------
// Kernel 4: proj GEMM (N=192) + bias, fused window reverse, fp32 output
// ---------------------------------------------------------------------------
__global__ __launch_bounds__(256) void proj_gemm(const _Float16* __restrict__ aw,
                                                 const float* __restrict__ Wg,
                                                 const float* __restrict__ bg,
                                                 float* __restrict__ out) {
  __shared__ __align__(16) _Float16 As[128 * 200];
  __shared__ __align__(16) _Float16 Bt[64 * 200];
  const int tid = threadIdx.x;
  const int m0 = blockIdx.x * 128;
  const int n0 = blockIdx.y * 64;

  for (int i = tid; i < 128 * 24; i += 256) {
    int m = i / 24, kc = i % 24;
    *(uint4*)(&As[m * 200 + kc * 8]) = *(const uint4*)(aw + (size_t)(m0 + m) * 192 + kc * 8);
  }
  for (int i = tid; i < 64 * 48; i += 256) {
    int n = i / 48, kc = i % 48;
    const float4 w4 = *(const float4*)(Wg + (size_t)(n0 + n) * 192 + kc * 4);
    half4_t h4 = { (_Float16)w4.x, (_Float16)w4.y, (_Float16)w4.z, (_Float16)w4.w };
    *(half4_t*)(&Bt[n * 200 + kc * 4]) = h4;
  }
  __syncthreads();

  const int wv = tid >> 5;
  const int lane = tid & 31;
  const int lh = lane & 15;
  const bool hiH = lane >= 16;
  const int mb = wv * 16;

  f32x8 acc[4] = {};
#pragma unroll
  for (int ks = 0; ks < 6; ++ks) {
    const _Float16* ap = &As[(mb + lh) * 200 + ks * 32 + (hiH ? 16 : 0)];
    half16_t aop = cat8(*(const half8_t*)ap, *(const half8_t*)(ap + 8));
#pragma unroll
    for (int ni = 0; ni < 4; ++ni) {
      const _Float16* bp = &Bt[(ni * 16 + lh) * 200 + ks * 32 + (hiH ? 16 : 0)];
      half16_t bop = cat8(*(const half8_t*)bp, *(const half8_t*)(bp + 8));
      acc[ni] = wmma_f16(aop, bop, acc[ni]);
    }
  }

#pragma unroll
  for (int ni = 0; ni < 4; ++ni) {
    int j = n0 + ni * 16 + lh;                       // output channel < 192
    float bj = bg[j];
#pragma unroll
    for (int r = 0; r < 8; ++r) {
      int m = m0 + mb + (hiH ? r + 8 : r);
      int w = m / 49;
      int p = m - w * 49;
      int b = w >> 6, win = w & 63;
      int wh = win >> 3, ww = win & 7;
      int pi = p / 7, pj = p - pi * 7;
      int norg = (wh * 7 + pi) * 56 + (ww * 7 + pj);
      out[((size_t)b * 3136 + norg) * 192 + j] = acc[ni][r] + bj;
    }
  }
}

// ---------------------------------------------------------------------------
extern "C" void kernel_launch(void* const* d_in, const int* in_sizes, int n_in,
                              void* d_out, int out_size, void* d_ws, size_t ws_size,
                              hipStream_t stream) {
  (void)in_sizes; (void)n_in; (void)out_size; (void)ws_size;
  const float* x      = (const float*)d_in[0];
  const float* gamma  = (const float*)d_in[1];
  const float* rpb    = (const float*)d_in[2];
  const float* qkv_w  = (const float*)d_in[3];
  const float* qkv_b  = (const float*)d_in[4];
  const float* proj_w = (const float*)d_in[5];
  const float* proj_b = (const float*)d_in[6];
  float* out = (float*)d_out;

  const size_t SEG = 38535168;                       // 100352*192*2 bytes
  char* ws = (char*)d_ws;
  _Float16* xw = (_Float16*)(ws);                    // LN'd tokens; reused as attn-out
  _Float16* qb = (_Float16*)(ws + SEG);
  _Float16* kb = (_Float16*)(ws + 2 * SEG);
  _Float16* vb = (_Float16*)(ws + 3 * SEG);
  _Float16* attn_out = xw;                           // xw dead after QKV GEMM

  hipLaunchKernelGGL(ln_partition, dim3(12544), dim3(256), 0, stream, x, gamma, xw);
  hipLaunchKernelGGL(qkv_gemm,  dim3(784, 9), dim3(256), 0, stream, xw, qkv_w, qkv_b, qb, kb, vb);
  hipLaunchKernelGGL(attn_win,  dim3(2048, 6), dim3(32), 0, stream, qb, kb, vb, rpb, attn_out);
  hipLaunchKernelGGL(proj_gemm, dim3(784, 3), dim3(256), 0, stream, attn_out, proj_w, proj_b, out);
}
